// SqueezeExcitation_54898271977731
// MI455X (gfx1250) — matrix operand order repaired and checked by reading, not compiled, not run on hardware
//
#include <hip/hip_runtime.h>
#include <hip/hip_bf16.h>

// Sigma-delta SNN block pair, algebraically fused for MI455X (gfx1250).
//   h   = qw1@x with t==0 shifted by -mean1            (fc1, Ind act)
//   out = diff(clip(cumsum_t(qw2@h), -6, 6))           (fc2, Clamp6)
// mean2 is provably zero; mean1 folds into a uniform prefix-sum offset.
// Cumsum/diff over T use DPP row_shr scans (VALU) instead of DS shuffles.

typedef _Float16 v16h __attribute__((ext_vector_type(16)));
typedef float    v8f  __attribute__((ext_vector_type(8)));

#define WMMA(A, B, Cacc) \
  __builtin_amdgcn_wmma_f32_16x16x32_f16(false, (A), false, (B), (short)0, (Cacc), false, false)

static constexpr int Nn   = 4;
static constexpr int Cin  = 128;
static constexpr int Cs   = 32;
static constexpr int Tt   = 32;
static constexpr int P    = 32 * 32 * 32;   // h*w*t contiguous per (n,c)
static constexpr int NP   = Nn * P;         // 131072 spatial columns
static constexpr int CT1  = NP / 16;        // 8192 column tiles (fc1)
static constexpr int PAIR = NP / 32;        // 4096 T-row pairs (fc2)

// ---------------------------------------------------- cross-lane helpers ----
// value of lane (l-N) within each 16-lane row; 0 when shifted past row start
template <int N>
__device__ __forceinline__ float rowshr(float v) {
#if __has_builtin(__builtin_amdgcn_update_dpp)
  return __builtin_bit_cast(float,
      __builtin_amdgcn_update_dpp(0, __builtin_bit_cast(int, v),
                                  0x110 | N, 0xF, 0xF, true));   // row_shr:N
#else
  float t = __shfl_up(v, (unsigned)N, 16);
  return ((int)(threadIdx.x & 15) >= N) ? t : 0.0f;
#endif
}

// broadcast lane 15 of each 16-lane row to the whole row
__device__ __forceinline__ float bcast15(float v) {
#if __has_builtin(__builtin_amdgcn_permlane16)
  unsigned i = __builtin_bit_cast(unsigned, v);
  return __builtin_bit_cast(float,
      __builtin_amdgcn_permlane16(i, i, 0xFFFFFFFFu, 0xFFFFFFFFu, false, false));
#else
  return __shfl(v, 15, 16);
#endif
}

// inclusive prefix sum within 16-lane rows (Hillis-Steele via DPP)
__device__ __forceinline__ float scan16(float v) {
  v += rowshr<1>(v);
  v += rowshr<2>(v);
  v += rowshr<4>(v);
  v += rowshr<8>(v);
  return v;
}

// ---------------------------------------------------------------- setup ----
__device__ __forceinline__ float quant8(float w) {
  // round(w/step)*step with step=1/32 (RNE like jnp.round), clamp [-2, 127/64]
  float q = rintf(w * 32.0f) * 0.03125f;
  return fminf(fmaxf(q, -2.0f), 1.984375f);
}

__global__ void sdnn_setup(const float* __restrict__ v1, const float* __restrict__ g1,
                           const float* __restrict__ v2, const float* __restrict__ g2,
                           _Float16* __restrict__ qw1, _Float16* __restrict__ qw2,
                           float* __restrict__ m1num) {
  const int tid = threadIdx.x;            // 128 threads, 1 block
  if (tid < Cs) m1num[tid] = 0.0f;        // zero mean1 accumulators every launch
  if (tid < Cs) {                         // qw1: [32 x 128] weight-norm + quantize
    float s = 0.0f;
    for (int i = 0; i < Cin; ++i) { float v = v1[tid * Cin + i]; s += v * v; }
    const float sc = g1[tid] / sqrtf(s);
    for (int i = 0; i < Cin; ++i)
      qw1[tid * Cin + i] = (_Float16)quant8(sc * v1[tid * Cin + i]);
  }
  if (tid < Cin) {                        // qw2: [128 x 32]
    float s = 0.0f;
    for (int i = 0; i < Cs; ++i) { float v = v2[tid * Cs + i]; s += v * v; }
    const float sc = g2[tid] / sqrtf(s);
    for (int i = 0; i < Cs; ++i)
      qw2[tid * Cs + i] = (_Float16)quant8(sc * v2[tid * Cs + i]);
  }
}

// ------------------------------------------------------------------ fc1 ----
// y1[n,o,q] = sum_i qw1[o,i] * x[n,i,q]   (o<32, i<128, q<P)
// plus per-channel reduction  m1num[o] += (T - t(q)) * y1
__global__ __launch_bounds__(256) void sdnn_fc1(const float* __restrict__ x,
                                                const _Float16* __restrict__ qw1,
                                                _Float16* __restrict__ y1,
                                                float* __restrict__ m1num,
                                                int total_waves) {
  __shared__ float part[8][Cs];           // per-wave partial channel sums

  const int lane = threadIdx.x & 31;
  const int l16  = lane & 15;             // A row / B,D column
  const int hi   = lane >> 4;
  const int wv   = threadIdx.x >> 5;
  const int wave = blockIdx.x * (blockDim.x >> 5) + wv;

  // A fragments: 2 M-tiles x 4 K-chunks, 16x32 f16 layout
  v16h A[2][4];
#pragma unroll
  for (int mt = 0; mt < 2; ++mt)
#pragma unroll
    for (int kc = 0; kc < 4; ++kc)
#pragma unroll
      for (int j = 0; j < 16; ++j) {
        const int K = 16 * hi + j;
        A[mt][kc][j] = qw1[(mt * 16 + l16) * Cin + kc * 32 + K];
      }

  float wacc[2][8];
#pragma unroll
  for (int mt = 0; mt < 2; ++mt)
#pragma unroll
    for (int r = 0; r < 8; ++r) wacc[mt][r] = 0.0f;

  for (int ct = wave; ct < CT1; ct += total_waves) {
    const int n  = ct >> 11;              // P/16 = 2048 tiles per n
    const int q0 = (ct & 2047) << 4;
    const float* xb = x + (size_t)n * Cin * P + q0 + l16;

    v8f c0 = {}; v8f c1 = {};
#pragma unroll
    for (int kc = 0; kc < 4; ++kc) {
      v16h b;                             // 32x16 f16 B: lane half picks K range
#pragma unroll
      for (int j = 0; j < 16; ++j) {
        const int ch = kc * 32 + hi * 16 + j;
        b[j] = (_Float16)xb[(size_t)ch * P];
      }
      c0 = WMMA(A[0][kc], b, c0);
      c1 = WMMA(A[1][kc], b, c1);
    }

    const float wt = (float)(Tt - ((q0 + l16) & (Tt - 1)));
    _Float16* yb = y1 + (size_t)n * Cs * P + q0 + l16;
#pragma unroll
    for (int r = 0; r < 8; ++r) {
      const int o0 = r + 8 * hi;
      const float v0 = c0[r];
      yb[(size_t)o0 * P] = (_Float16)v0;
      wacc[0][r] += v0 * wt;
      const int o1 = 16 + r + 8 * hi;
      const float v1 = c1[r];
      yb[(size_t)o1 * P] = (_Float16)v1;
      wacc[1][r] += v1 * wt;
    }
  }

  // 16-lane tree reduce per half-row, stash per-wave partials in LDS,
  // then one atomic per channel per BLOCK (avoids same-address atomic storms)
#pragma unroll
  for (int mt = 0; mt < 2; ++mt)
#pragma unroll
    for (int r = 0; r < 8; ++r) {
      float s = wacc[mt][r];
#pragma unroll
      for (int d = 1; d < 16; d <<= 1) s += __shfl_xor(s, d, 32);
      if (l16 == 0) part[wv][mt * 16 + r + 8 * hi] = s;
    }
  __syncthreads();
  if (threadIdx.x < Cs) {
    float s = 0.0f;
#pragma unroll
    for (int w = 0; w < 8; ++w) s += part[w][threadIdx.x];
    atomicAdd(&m1num[threadIdx.x], s);
  }
}

// ------------------------------------------------------------------ fc2 ----
// raw = qw2 @ y1 ; s2 = cumsum_t(raw) - corr[c]  (corr = qw2 @ mean1, since a
// t==0 delta shifts every prefix sum) ; out = diff(clip(s2, -6, 6)).
__global__ __launch_bounds__(256) void sdnn_fc2(const _Float16* __restrict__ y1,
                                                const _Float16* __restrict__ qw2,
                                                const float* __restrict__ m1num,
                                                float* __restrict__ out,
                                                int total_waves) {
  __shared__ float corr[Cin];
  if (threadIdx.x < Cin) {                // corr[c] = sum_o qw2[c,o] * mean1[o]
    float s = 0.0f;
#pragma unroll
    for (int i = 0; i < Cs; ++i)
      s += (float)qw2[threadIdx.x * Cs + i] * (m1num[i] * (1.0f / (float)NP));
    corr[threadIdx.x] = s;
  }
  __syncthreads();

  const int lane = threadIdx.x & 31;
  const int l16  = lane & 15;
  const int hi   = lane >> 4;
  const int wave = blockIdx.x * (blockDim.x >> 5) + (threadIdx.x >> 5);

  v16h A[8];                              // 8 M-tiles of qw2 (K = 32, one chunk)
#pragma unroll
  for (int mt = 0; mt < 8; ++mt)
#pragma unroll
    for (int j = 0; j < 16; ++j) {
      const int K = 16 * hi + j;
      A[mt][j] = qw2[(mt * 16 + l16) * Cs + K];
    }

  for (int pt = wave; pt < PAIR; pt += total_waves) {
    const int n  = pt >> 10;              // P/32 = 1024 pairs per n
    const int q0 = (pt & 1023) << 5;      // 32-aligned -> cols = t = 0..31
    const _Float16* hb = y1 + (size_t)n * Cs * P + q0 + l16;

    v16h b0, b1;                          // cols t=0..15 and t=16..31
#pragma unroll
    for (int j = 0; j < 16; ++j) {
      const int ch = hi * 16 + j;
      b0[j] = hb[(size_t)ch * P];
      b1[j] = hb[(size_t)ch * P + 16];
    }

    float* ob = out + (size_t)n * Cin * P + q0 + l16;
#pragma unroll
    for (int mt = 0; mt < 8; ++mt) {
      v8f d0 = {}; v8f d1 = {};
      d0 = WMMA(A[mt], b0, d0);
      d1 = WMMA(A[mt], b1, d1);
#pragma unroll
      for (int r = 0; r < 8; ++r) {
        const int o = mt * 16 + r + 8 * hi;
        // raw inclusive prefix sums over t (DPP scans, no DS traffic)
        float s0 = scan16(d0[r]);
        float s1 = scan16(d1[r]) + bcast15(s0);      // carry t<=15 total
        const float cv = corr[o];                    // uniform per half-row
        const float a0 = fminf(fmaxf(s0 - cv, -6.0f), 6.0f);   // Clamp6
        const float a1 = fminf(fmaxf(s1 - cv, -6.0f), 6.0f);
        float p0 = rowshr<1>(a0);                    // 0 at t=0 -> d[0]=a[0]
        float p1 = rowshr<1>(a1);
        const float a015 = bcast15(a0);
        if (l16 == 0) p1 = a015;                     // t=16 prev is a[15]
        ob[(size_t)o * P]      = a0 - p0;
        ob[(size_t)o * P + 16] = a1 - p1;
      }
    }
  }
}

// ------------------------------------------------------------- launcher ----
extern "C" void kernel_launch(void* const* d_in, const int* in_sizes, int n_in,
                              void* d_out, int out_size, void* d_ws, size_t ws_size,
                              hipStream_t stream) {
  const float* x  = (const float*)d_in[0];
  const float* v1 = (const float*)d_in[1];
  const float* g1 = (const float*)d_in[2];
  const float* v2 = (const float*)d_in[3];
  const float* g2 = (const float*)d_in[4];
  float* out = (float*)d_out;

  char* ws = (char*)d_ws;
  _Float16* qw1  = (_Float16*)(ws);            //  8 KB  [32x128]
  _Float16* qw2  = (_Float16*)(ws + 8192);     //  8 KB  [128x32]
  float*    m1n  = (float*)   (ws + 16384);    //  128 B [32]
  _Float16* y1ws = (_Float16*)(ws + 32768);    //  8 MB  [4x32x32768] f16

  hipLaunchKernelGGL(sdnn_setup, dim3(1), dim3(128), 0, stream,
                     v1, g1, v2, g2, qw1, qw2, m1n);

  const int blk1 = 256;                        // 2048 waves -> 4 tiles each
  hipLaunchKernelGGL(sdnn_fc1, dim3(blk1), dim3(256), 0, stream,
                     x, qw1, y1ws, m1n, blk1 * 8);

  const int blk2 = 256;                        // 2048 waves -> 2 pairs each
  hipLaunchKernelGGL(sdnn_fc2, dim3(blk2), dim3(256), 0, stream,
                     y1ws, qw2, m1n, out, blk2 * 8);
}
